// GemNetTEwald_57904749085213
// MI455X (gfx1250) — matrix operand order repaired and checked by reading, not compiled, not run on hardware
//
#include <hip/hip_runtime.h>

// ---------------- dimensions ----------------
#define NA    16384      // atoms
#define BMOL  64         // molecules
#define NE    262144     // edges
#define NT3   1048576    // triplets
#define RDIM  128
#define SDIM  7
#define ADIM  128
#define EMDIM 256
#define TRDIM 32
#define CRBF  16
#define CDIM  16
#define BILD  64
#define LNUM  2
#define KPTS  32
#define APM   256        // atoms per molecule
#define CUT   6.0f

typedef __attribute__((ext_vector_type(16))) __bf16 v16bf;
typedef __attribute__((ext_vector_type(8)))  __bf16 v8bf;
typedef __attribute__((ext_vector_type(8)))  float  v8f;

__device__ __forceinline__ float silu_f(float x) { return x / (1.0f + __expf(-x)); }

__device__ __forceinline__ void atomAddF(float* p, float v) {
  __hip_atomic_fetch_add(p, v, __ATOMIC_RELAXED, __HIP_MEMORY_SCOPE_AGENT);
}

// ---- WMMA fragment helpers (wave32) ----
// A (16x32 bf16, row-major source, per ISA table):
//   lanes 0-15 : row=lane,    K = {0..7, 16..23}
//   lanes 16-31: row=lane-16, K = {8..15, 24..31}
__device__ __forceinline__ v16bf fragA(const __bf16* base, int stride) {
  int lane = threadIdx.x & 31;
  const __bf16* p = base + (lane & 15) * stride + ((lane >> 4) << 3);
  v8bf lo = *(const v8bf*)(p);
  v8bf hi = *(const v8bf*)(p + 16);
  return __builtin_shufflevector(lo, hi, 0,1,2,3,4,5,6,7,8,9,10,11,12,13,14,15);
}
// B (32x16) from Bt layout [N][K]: lanes 0-15: col=lane, K={0..7,16..23}; lanes 16-31: col=lane-16, K={8..15,24..31}
__device__ __forceinline__ v16bf fragB(const __bf16* base, int stride) {
  int lane = threadIdx.x & 31;
  const __bf16* p = base + (lane & 15) * stride + ((lane >> 4) << 3);
  v8bf lo = *(const v8bf*)(p);
  v8bf hi = *(const v8bf*)(p + 16);
  return __builtin_shufflevector(lo, hi, 0,1,2,3,4,5,6,7,8,9,10,11,12,13,14,15);
}
__device__ __forceinline__ v8f wmma_bf(v16bf a, v16bf b, v8f c) {
  return __builtin_amdgcn_wmma_f32_16x16x32_bf16(false, a, false, b, (short)0, c, false, false);
}

// ---------------- utility kernels ----------------
__global__ void k_zero(float* __restrict__ p, long n) {
  long i = (long)blockIdx.x * blockDim.x + threadIdx.x;
  if (i < n) p[i] = 0.0f;
}
__global__ void k_f2b(const float* __restrict__ s, __bf16* __restrict__ d, long n) {
  long i = (long)blockIdx.x * blockDim.x + threadIdx.x;
  if (i < n) d[i] = (__bf16)s[i];
}
// W[K][N] f32  ->  Bt[N][K] bf16
__global__ void k_wt(const float* __restrict__ W, __bf16* __restrict__ Bt, int K, int N) {
  long i = (long)blockIdx.x * blockDim.x + threadIdx.x;
  if (i >= (long)K * N) return;
  int n = (int)(i / K), k = (int)(i % K);
  Bt[i] = (__bf16)W[(long)k * N + n];
}
// tiny f32 GEMM (fused rbf weights)
__global__ void k_ngemm(const float* __restrict__ A, const float* __restrict__ Bw,
                        float* __restrict__ C, int M, int N, int K) {
  long i = (long)blockIdx.x * blockDim.x + threadIdx.x;
  if (i >= (long)M * N) return;
  int m = (int)(i / N), n = (int)(i % N);
  float acc = 0.0f;
  for (int k = 0; k < K; ++k) acc += A[(long)m * K + k] * Bw[(long)k * N + n];
  C[i] = acc;
}

// ---------------- model kernels ----------------
__global__ void k_atom(const float* __restrict__ tab, const int* __restrict__ z,
                       float* __restrict__ h, __bf16* __restrict__ hbf) {
  long i = (long)blockIdx.x * blockDim.x + threadIdx.x;
  if (i >= (long)NA * ADIM) return;
  int n = (int)(i >> 7), a = (int)(i & 127);
  float v = tab[(long)z[n] * ADIM + a];
  h[i] = v; hbf[i] = (__bf16)v;
}

__global__ __launch_bounds__(128)
void k_geom(const float* __restrict__ pos, const int* __restrict__ src, const int* __restrict__ tgt,
            float* __restrict__ vec, float* __restrict__ dd,
            float* __restrict__ rbf, __bf16* __restrict__ rbfb) {
  int e = blockIdx.x;
  int r = threadIdx.x;
  int s = src[e], t = tgt[e];
  float vx = pos[(long)s*3+0] - pos[(long)t*3+0];
  float vy = pos[(long)s*3+1] - pos[(long)t*3+1];
  float vz = pos[(long)s*3+2] - pos[(long)t*3+2];
  float d = sqrtf(vx*vx + vy*vy + vz*vz + 1e-12f);
  if (r == 0) { vec[(long)e*3+0]=vx; vec[(long)e*3+1]=vy; vec[(long)e*3+2]=vz; dd[e]=d; }
  float ds = d * (1.0f / CUT);
  float d2 = ds*ds;
  float d5 = d2*d2*ds;
  float env = (ds < 1.0f) ? (1.0f + d5*(-21.0f + ds*(35.0f - 15.0f*ds))) : 0.0f;
  float freq = (float)(r + 1) * 3.14159265358979323846f;
  float v = env * 0.5773502691896258f * __sinf(freq * ds) / d;
  rbf[(long)e*RDIM + r] = v;
  rbfb[(long)e*RDIM + r] = (__bf16)v;
}

__global__ void k_kdot(const float* __restrict__ pos, const float* __restrict__ kg,
                       float* __restrict__ ck, float* __restrict__ sk) {
  long i = (long)blockIdx.x * blockDim.x + threadIdx.x;
  if (i >= (long)NA * KPTS) return;
  int n = (int)(i >> 5), k = (int)(i & 31);
  float dp = pos[(long)n*3+0]*kg[(long)k*3+0] + pos[(long)n*3+1]*kg[(long)k*3+1]
           + pos[(long)n*3+2]*kg[(long)k*3+2];
  float s, c; __sincosf(dp, &s, &c);
  ck[i] = c; sk[i] = s;
}

__global__ __launch_bounds__(128)
void k_kf(const float* __restrict__ krbf, const float* __restrict__ Wd,
          const float* __restrict__ Wkf, float* __restrict__ kf) {
  long i = (long)blockIdx.x * blockDim.x + threadIdx.x;
  if (i >= (long)KPTS * ADIM) return;
  int k = (int)(i >> 7), a = (int)(i & 127);
  float acc = 0.0f;
  for (int dp = 0; dp < 8; ++dp) {
    float t = 0.0f;
    for (int r = 0; r < 128; ++r) t += krbf[(long)k*128 + r] * Wd[(long)r*8 + dp];
    acc += silu_f(t) * Wkf[(long)dp*ADIM + a];
  }
  kf[i] = acc;
}

// rW[e, s*16+c] = sum_r rbf[e,r] * W_cbf3[s,r,c]
__global__ __launch_bounds__(128)
void k_rw(const float* __restrict__ rbf, const float* __restrict__ Wcbf, float* __restrict__ rW) {
  __shared__ float sr[RDIM];
  int e = blockIdx.x;
  sr[threadIdx.x] = rbf[(long)e * RDIM + threadIdx.x];
  __syncthreads();
  int t = threadIdx.x;
  if (t < SDIM * CDIM) {
    int s = t >> 4, c = t & 15;
    float acc = 0.0f;
    for (int r = 0; r < RDIM; ++r) acc += sr[r] * Wcbf[((long)s * RDIM + r) * CDIM + c];
    rW[(long)e * (SDIM * CDIM) + t] = acc;
  }
}

__global__ __launch_bounds__(256)
void k_cbf(const float* __restrict__ vec, const float* __restrict__ dd,
           const float* __restrict__ rW, const int* __restrict__ ba,
           const int* __restrict__ ca, float* __restrict__ cbf) {
  long t = (long)blockIdx.x * blockDim.x + threadIdx.x;
  if (t >= (long)NT3) return;
  int eb = ba[t], ec = ca[t];
  float dot = vec[(long)eb*3]*vec[(long)ec*3] + vec[(long)eb*3+1]*vec[(long)ec*3+1]
            + vec[(long)eb*3+2]*vec[(long)ec*3+2];
  float x = dot / (dd[eb] * dd[ec]);
  x = fminf(1.0f, fmaxf(-1.0f, x));
  float P[SDIM];
  P[0] = 1.0f; P[1] = x;
  #pragma unroll
  for (int l = 1; l < SDIM - 1; ++l)
    P[l+1] = ((float)(2*l+1) * x * P[l] - (float)l * P[l-1]) / (float)(l+1);
  const float* rwr = rW + (long)ec * (SDIM * CDIM);
  #pragma unroll
  for (int c = 0; c < CDIM; ++c) {
    float acc = 0.0f;
    #pragma unroll
    for (int s = 0; s < SDIM; ++s) acc += P[s] * rwr[s * CDIM + c];
    cbf[t * CDIM + c] = acc;
  }
}

__global__ void k_concat(const __bf16* __restrict__ hbf, const __bf16* __restrict__ third,
                         const int* __restrict__ src, const int* __restrict__ tgt,
                         __bf16* __restrict__ out, int W3) {
  long i = (long)blockIdx.x * blockDim.x + threadIdx.x;
  int Wtot = 2 * ADIM + W3;
  long e = i / Wtot;
  if (e >= (long)NE) return;
  int j = (int)(i % Wtot);
  __bf16 v;
  if (j < ADIM)            v = hbf[(long)src[e] * ADIM + j];
  else if (j < 2 * ADIM)   v = hbf[(long)tgt[e] * ADIM + (j - ADIM)];
  else                     v = third[(long)e * W3 + (j - 2 * ADIM)];
  out[i] = v;
}

// ---------------- main WMMA GEMM ----------------
// C[M, Nfull] = A[M,K](bf16) x Bt[Nfull,K](bf16 pre-transposed), fused epilogue.
// Block = 4 waves; wave w owns rows row0..row0+15, block covers NT*16 cols at blockIdx.y.
// B chunk staged in LDS once per block per k-step (shared by all 4 waves).
// flags: 1=silu, 2=mul by emul[M,Nfull], 4=resid add+scale
template<int NT>
__global__ __launch_bounds__(128)
void k_gemm(const __bf16* __restrict__ A, const __bf16* __restrict__ Bt,
            const float* __restrict__ bias, const float* __restrict__ emul,
            const float* __restrict__ resid, float rscale,
            float* __restrict__ Cf, __bf16* __restrict__ Cb,
            int K, int Nfull, int flags) {
  __shared__ __align__(16) __bf16 sB[NT * 16 * 32];
  int wave = threadIdx.x >> 5;
  long row0 = ((long)blockIdx.x * 4 + wave) * 16;
  int ncol0 = blockIdx.y * (NT * 16);
  v8f acc[NT];
  #pragma unroll
  for (int i = 0; i < NT; ++i)
    #pragma unroll
    for (int j = 0; j < 8; ++j) acc[i][j] = 0.0f;
  const __bf16* pa = A + row0 * K;
  const __bf16* pb = Bt + (long)ncol0 * K;
  for (int k0 = 0; k0 < K; k0 += 32) {
    __syncthreads();
    #pragma unroll
    for (int ch = 0; ch < NT * 16 * 4 / 128; ++ch) {
      int c = ch * 128 + threadIdx.x;
      int n = c >> 2, kg = (c & 3) << 3;
      *(v8bf*)(sB + n * 32 + kg) = *(const v8bf*)(pb + (long)n * K + k0 + kg);
    }
    __syncthreads();
    v16bf a = fragA(pa + k0, K);
    __builtin_prefetch(pa + k0 + 32, 0, 1);
    #pragma unroll
    for (int nt = 0; nt < NT; ++nt) {
      v16bf b = fragB(sB + nt * 16 * 32, 32);
      acc[nt] = wmma_bf(a, b, acc[nt]);
    }
  }
  int lane = threadIdx.x & 31;
  int nloc = lane & 15;
  int mbase = (lane >> 4) << 3;
  #pragma unroll
  for (int nt = 0; nt < NT; ++nt) {
    #pragma unroll
    for (int v = 0; v < 8; ++v) {
      long m = row0 + mbase + v;
      int n = ncol0 + nt * 16 + nloc;
      long idx = m * (long)Nfull + n;
      float val = acc[nt][v];
      if (bias) val += bias[n];
      if (flags & 1) val = silu_f(val);
      if (flags & 2) val *= emul[idx];
      if (flags & 4) val = (resid[idx] + val) * rscale;
      if (Cf) Cf[idx] = val;
      if (Cb) Cb[idx] = (__bf16)val;
    }
  }
}

// ---------------- triplet bilinear (WMMA + atomic scatter) ----------------
// per 16 triplets: z = sum_c diag(cbf[:,c]) * (Xkj @ Wbil[c]);  x3[id3_ca] += z
// W_bil staged through LDS in two 8-channel phases (32KB), shared by 8 waves.
__global__ __launch_bounds__(256)
void k_trip(const __bf16* __restrict__ xtrip, const float* __restrict__ cbf,
            const int* __restrict__ ba, const int* __restrict__ ca,
            const __bf16* __restrict__ Wbil /* [16][64][32] bf16 (Bt per channel) */,
            float* __restrict__ x3) {
  __shared__ __align__(16) __bf16 sW[8 * BILD * TRDIM];   // 16384 halfs = 32KB
  int wave = threadIdx.x >> 5;
  long t0 = ((long)blockIdx.x * 8 + wave) * 16;
  int lane = threadIdx.x & 31;
  int r = lane & 15;
  long trow = t0 + r;
  int ekj = ba[trow];
  const __bf16* xrow = xtrip + (long)ekj * TRDIM;
  int kb = (lane >> 4) << 3;
  v8bf lo = *(const v8bf*)(xrow + kb);
  v8bf hi = *(const v8bf*)(xrow + kb + 16);
  v16bf araw = __builtin_shufflevector(lo, hi, 0,1,2,3,4,5,6,7,8,9,10,11,12,13,14,15);
  v8f acc[4];
  #pragma unroll
  for (int i = 0; i < 4; ++i)
    #pragma unroll
    for (int j = 0; j < 8; ++j) acc[i][j] = 0.0f;
  for (int ph = 0; ph < 2; ++ph) {
    __syncthreads();
    #pragma unroll
    for (int it = 0; it < 8; ++it) {
      int ch = it * 256 + threadIdx.x;        // 2048 chunks of 8 halfs
      *(v8bf*)(sW + ch * 8) = *(const v8bf*)(Wbil + (long)ph * 8 * BILD * TRDIM + ch * 8);
    }
    __syncthreads();
    for (int c = 0; c < 8; ++c) {
      float s = cbf[trow * CDIM + ph * 8 + c];
      v16bf a;
      #pragma unroll
      for (int h = 0; h < 16; ++h) a[h] = (__bf16)((float)araw[h] * s);
      const __bf16* Bc = sW + c * BILD * TRDIM;
      #pragma unroll
      for (int nt = 0; nt < 4; ++nt) {
        v16bf b = fragB(Bc + (long)(nt * 16) * TRDIM, TRDIM);
        acc[nt] = wmma_bf(a, b, acc[nt]);
      }
    }
  }
  int nloc = lane & 15;
  int mbase = (lane >> 4) << 3;
  #pragma unroll
  for (int nt = 0; nt < 4; ++nt)
    #pragma unroll
    for (int v = 0; v < 8; ++v) {
      long t = t0 + mbase + v;
      int e = ca[t];
      atomAddF(x3 + (long)e * BILD + nt * 16 + nloc, acc[nt][v]);
    }
}

// edge->atom scatter: out[tgt[e], c] += m[e,c]*g[e,c]
__global__ void k_scatter(const float* __restrict__ m, const float* __restrict__ g,
                          const int* __restrict__ tgt, float* __restrict__ out) {
  long i = (long)blockIdx.x * blockDim.x + threadIdx.x;
  if (i >= (long)NE * EMDIM) return;
  long e = i >> 8; int c = (int)(i & 255);
  atomAddF(out + (long)tgt[e] * EMDIM + c, m[i] * g[i]);
}

// Ewald structure factors: sf[b,k,:] = kf[k,:] * sum_{n in mol b} {cos,sin}k[n,k]*h[n,:]
__global__ __launch_bounds__(128)
void k_sf(const float* __restrict__ ck, const float* __restrict__ sk,
          const float* __restrict__ h, const float* __restrict__ kf,
          float* __restrict__ sfre, float* __restrict__ sfim) {
  int b = blockIdx.x >> 5;
  int k = blockIdx.x & 31;
  int a = threadIdx.x;
  float ar = 0.0f, ai = 0.0f;
  int nb = b * APM;
  for (int i = 0; i < APM; ++i) {
    float hv = h[(long)(nb + i) * ADIM + a];
    ar += ck[(long)(nb + i) * KPTS + k] * hv;
    ai += sk[(long)(nb + i) * KPTS + k] * hv;
  }
  float kfv = kf[(long)k * ADIM + a];
  long o = ((long)b * KPTS + k) * ADIM + a;
  sfre[o] = ar * kfv;
  sfim[o] = ai * kfv;
}

__global__ __launch_bounds__(128)
void k_ew(const float* __restrict__ ck, const float* __restrict__ sk,
          const float* __restrict__ sfre, const float* __restrict__ sfim,
          float* __restrict__ ew) {
  int n = blockIdx.x;
  int a = threadIdx.x;
  int b = n / APM;
  float acc = 0.0f;
  for (int k = 0; k < KPTS; ++k) {
    long o = ((long)b * KPTS + k) * ADIM + a;
    acc += ck[(long)n * KPTS + k] * sfre[o] + sk[(long)n * KPTS + k] * sfim[o];
  }
  ew[(long)n * ADIM + a] = acc;
}

__global__ void k_hup(const float* __restrict__ ha, const float* __restrict__ ewW,
                      float* __restrict__ h, __bf16* __restrict__ hbf) {
  long i = (long)blockIdx.x * blockDim.x + threadIdx.x;
  if (i >= (long)NA * ADIM) return;
  float v = (h[i] + silu_f(ha[i]) + silu_f(ewW[i])) * 0.5773502691896258f;
  h[i] = v; hbf[i] = (__bf16)v;
}

// per-molecule energy: out[b] = sum_{atoms} q[n,:].W2
__global__ __launch_bounds__(256)
void k_final(const float* __restrict__ q, const float* __restrict__ w2, float* __restrict__ out) {
  __shared__ float sd[256];
  int b = blockIdx.x, t = threadIdx.x;
  long n = (long)b * APM + t;
  float acc = 0.0f;
  for (int a = 0; a < ADIM; ++a) acc += q[n * ADIM + a] * w2[a];
  sd[t] = acc;
  __syncthreads();
  for (int s = 128; s > 0; s >>= 1) {
    if (t < s) sd[t] += sd[t + s];
    __syncthreads();
  }
  if (t == 0) out[b] = sd[0];
}

// ---------------- host orchestration ----------------
static void gemm_launch(hipStream_t st, const __bf16* A, const __bf16* Bt,
                        const float* bias, const float* emul, const float* resid,
                        float rscale, float* Cf, __bf16* Cb, long M, int Nn, int K, int flags) {
  int tiles = Nn / 16;
  dim3 b(128);
  if (tiles % 8 == 0) {
    dim3 g((unsigned)(M / 64), (unsigned)(tiles / 8));
    k_gemm<8><<<g, b, 0, st>>>(A, Bt, bias, emul, resid, rscale, Cf, Cb, K, Nn, flags);
  } else if (tiles == 4) {
    dim3 g((unsigned)(M / 64), 1);
    k_gemm<4><<<g, b, 0, st>>>(A, Bt, bias, emul, resid, rscale, Cf, Cb, K, Nn, flags);
  } else if (tiles == 2) {
    dim3 g((unsigned)(M / 64), 1);
    k_gemm<2><<<g, b, 0, st>>>(A, Bt, bias, emul, resid, rscale, Cf, Cb, K, Nn, flags);
  } else {
    dim3 g((unsigned)(M / 64), 1);
    k_gemm<1><<<g, b, 0, st>>>(A, Bt, bias, emul, resid, rscale, Cf, Cb, K, Nn, flags);
  }
}

extern "C" void kernel_launch(void* const* d_in, const int* in_sizes, int n_in,
                              void* d_out, int out_size, void* d_ws, size_t ws_size,
                              hipStream_t stream) {
  (void)in_sizes; (void)n_in; (void)out_size; (void)ws_size;
  const float* pos      = (const float*)d_in[0];
  const float* atab     = (const float*)d_in[1];
  const float* W_edge   = (const float*)d_in[2];
  const float* b_edge   = (const float*)d_in[3];
  const float* W_rbf3   = (const float*)d_in[4];
  const float* W_rbf3_E = (const float*)d_in[5];
  const float* W_cbf3   = (const float*)d_in[6];
  const float* W_ba     = (const float*)d_in[7];
  const float* W_dt     = (const float*)d_in[8];
  const float* W_bil    = (const float*)d_in[9];
  const float* W_up     = (const float*)d_in[10];
  const float* W_r1     = (const float*)d_in[11];
  const float* W_r2     = (const float*)d_in[12];
  const float* W_rbf_h  = (const float*)d_in[13];
  const float* W_h      = (const float*)d_in[14];
  const float* W_atom   = (const float*)d_in[15];
  const float* k_grid   = (const float*)d_in[16];
  const float* k_rbf    = (const float*)d_in[17];
  const float* W_downk  = (const float*)d_in[18];
  const float* W_kfilt  = (const float*)d_in[19];
  const float* W_ew     = (const float*)d_in[20];
  const float* W_cat    = (const float*)d_in[21];
  const float* W_rbf_o  = (const float*)d_in[22];
  const float* W_oute   = (const float*)d_in[23];
  const float* W_out1   = (const float*)d_in[24];
  const float* W_out2   = (const float*)d_in[25];
  const int*   z        = (const int*)d_in[26];
  const int*   e_src    = (const int*)d_in[27];
  const int*   e_tgt    = (const int*)d_in[28];
  const int*   id_ba    = (const int*)d_in[29];
  const int*   id_ca    = (const int*)d_in[30];
  float* out = (float*)d_out;

  char* wsb = (char*)d_ws;
  size_t off = 0;
  auto alloc = [&](size_t bytes) -> char* {
    char* p = wsb + off;
    off = (off + bytes + 255) & ~(size_t)255;
    return p;
  };
  // f32 buffers
  float* vecb  = (float*)alloc((size_t)NE * 3 * 4);
  float* ddb   = (float*)alloc((size_t)NE * 4);
  float* rbff  = (float*)alloc((size_t)NE * RDIM * 4);
  float* hf    = (float*)alloc((size_t)NA * ADIM * 4);
  float* mf    = (float*)alloc((size_t)NE * EMDIM * 4);
  float* rWb   = (float*)alloc((size_t)NE * SDIM * CDIM * 4);
  float* cbfb  = (float*)alloc((size_t)NT3 * CDIM * 4);
  float* x3f   = (float*)alloc((size_t)NE * BILD * 4);
  float* gate  = (float*)alloc((size_t)NE * EMDIM * 4);   // rbf3e / g / out-gate
  float* Hm    = (float*)alloc((size_t)NA * EMDIM * 4);   // also Ea
  float* haf   = (float*)alloc((size_t)NA * ADIM * 4);
  float* cosk  = (float*)alloc((size_t)NA * KPTS * 4);
  float* sink  = (float*)alloc((size_t)NA * KPTS * 4);
  float* sfre  = (float*)alloc((size_t)BMOL * KPTS * ADIM * 4);
  float* sfim  = (float*)alloc((size_t)BMOL * KPTS * ADIM * 4);
  float* kff   = (float*)alloc((size_t)KPTS * ADIM * 4);
  float* ewf   = (float*)alloc((size_t)NA * ADIM * 4);
  float* ewWf  = (float*)alloc((size_t)NA * ADIM * 4);
  float* qf    = (float*)alloc((size_t)NA * ADIM * 4);
  float* ftmp  = (float*)alloc((size_t)RDIM * EMDIM * 4);
  // bf16 buffers
  __bf16* rbfb  = (__bf16*)alloc((size_t)NE * RDIM * 2);
  __bf16* hbf   = (__bf16*)alloc((size_t)NA * ADIM * 2);
  __bf16* mb    = (__bf16*)alloc((size_t)NE * EMDIM * 2);
  __bf16* xtb   = (__bf16*)alloc((size_t)NE * TRDIM * 2);
  __bf16* x3b   = (__bf16*)alloc((size_t)NE * BILD * 2);
  __bf16* X1b   = (__bf16*)alloc((size_t)NE * EMDIM * 2);
  __bf16* Hmb   = (__bf16*)alloc((size_t)NA * EMDIM * 2);
  __bf16* ewb   = (__bf16*)alloc((size_t)NA * ADIM * 2);
  __bf16* Acat  = (__bf16*)alloc((size_t)NE * 512 * 2);
  // transposed bf16 weights
  __bf16* wtEdge = (__bf16*)alloc((size_t)256 * 384 * 2);
  __bf16* wtBa   = (__bf16*)alloc((size_t)LNUM * 256 * 256 * 2);
  __bf16* wtDt   = (__bf16*)alloc((size_t)LNUM * 32 * 256 * 2);
  __bf16* wtBil  = (__bf16*)alloc((size_t)LNUM * CDIM * BILD * TRDIM * 2);
  __bf16* wtUp   = (__bf16*)alloc((size_t)LNUM * 256 * 64 * 2);
  __bf16* wtR1   = (__bf16*)alloc((size_t)LNUM * 256 * 256 * 2);
  __bf16* wtR2   = (__bf16*)alloc((size_t)LNUM * 256 * 256 * 2);
  __bf16* wtAt   = (__bf16*)alloc((size_t)LNUM * 128 * 256 * 2);
  __bf16* wtEwW  = (__bf16*)alloc((size_t)LNUM * 128 * 128 * 2);
  __bf16* wtCat  = (__bf16*)alloc((size_t)LNUM * 256 * 512 * 2);
  __bf16* wtO1   = (__bf16*)alloc((size_t)128 * 256 * 2);
  __bf16* wtF3   = (__bf16*)alloc((size_t)LNUM * 256 * 128 * 2);
  __bf16* wtFh   = (__bf16*)alloc((size_t)LNUM * 256 * 128 * 2);
  __bf16* wtFo   = (__bf16*)alloc((size_t)256 * 128 * 2);

  const float isq2 = 0.70710678118654752f;
  auto nblk = [](long n, int b) { return (unsigned)((n + b - 1) / b); };

  // ---- weight prep ----
  k_wt<<<nblk((long)384 * 256, 256), 256, 0, stream>>>(W_edge, wtEdge, 384, 256);
  k_wt<<<nblk((long)256 * 128, 256), 256, 0, stream>>>(W_out1, wtO1, 256, 128);
  for (int l = 0; l < LNUM; ++l) {
    k_wt<<<nblk(256 * 256, 256), 256, 0, stream>>>(W_ba + (long)l * 256 * 256, wtBa + (long)l * 256 * 256, 256, 256);
    k_wt<<<nblk(256 * 32, 256), 256, 0, stream>>>(W_dt + (long)l * 256 * 32, wtDt + (long)l * 32 * 256, 256, 32);
    k_wt<<<nblk(64 * 256, 256), 256, 0, stream>>>(W_up + (long)l * 64 * 256, wtUp + (long)l * 256 * 64, 64, 256);
    k_wt<<<nblk(256 * 256, 256), 256, 0, stream>>>(W_r1 + (long)l * 256 * 256, wtR1 + (long)l * 256 * 256, 256, 256);
    k_wt<<<nblk(256 * 256, 256), 256, 0, stream>>>(W_r2 + (long)l * 256 * 256, wtR2 + (long)l * 256 * 256, 256, 256);
    k_wt<<<nblk(256 * 128, 256), 256, 0, stream>>>(W_atom + (long)l * 256 * 128, wtAt + (long)l * 128 * 256, 256, 128);
    k_wt<<<nblk(128 * 128, 256), 256, 0, stream>>>(W_ew + (long)l * 128 * 128, wtEwW + (long)l * 128 * 128, 128, 128);
    k_wt<<<nblk(512 * 256, 256), 256, 0, stream>>>(W_cat + (long)l * 512 * 256, wtCat + (long)l * 256 * 512, 512, 256);
    for (int c = 0; c < CDIM; ++c)
      k_wt<<<nblk(32 * 64, 256), 256, 0, stream>>>(W_bil + ((long)l * CDIM + c) * 32 * 64,
                                                   wtBil + ((long)l * CDIM + c) * 64 * 32, 32, 64);
    // fused rbf weights (f32 small gemm -> transpose to bf16)
    k_ngemm<<<nblk(128 * 256, 256), 256, 0, stream>>>(W_rbf3, W_rbf3_E + (long)l * 16 * 256, ftmp, 128, 256, 16);
    k_wt<<<nblk(128 * 256, 256), 256, 0, stream>>>(ftmp, wtF3 + (long)l * 256 * 128, 128, 256);
    k_ngemm<<<nblk(128 * 256, 256), 256, 0, stream>>>(W_rbf_h, W_h + (long)l * 16 * 256, ftmp, 128, 256, 16);
    k_wt<<<nblk(128 * 256, 256), 256, 0, stream>>>(ftmp, wtFh + (long)l * 256 * 128, 128, 256);
  }
  k_ngemm<<<nblk(128 * 256, 256), 256, 0, stream>>>(W_rbf_o, W_oute, ftmp, 128, 256, 16);
  k_wt<<<nblk(128 * 256, 256), 256, 0, stream>>>(ftmp, wtFo, 128, 256);

  // ---- geometry / bases / embeddings ----
  k_atom<<<nblk((long)NA * ADIM, 256), 256, 0, stream>>>(atab, z, hf, hbf);
  k_geom<<<NE, 128, 0, stream>>>(pos, e_src, e_tgt, vecb, ddb, rbff, rbfb);
  k_kdot<<<nblk((long)NA * KPTS, 256), 256, 0, stream>>>(pos, k_grid, cosk, sink);
  k_kf<<<nblk((long)KPTS * ADIM, 128), 128, 0, stream>>>(k_rbf, W_downk, W_kfilt, kff);
  k_rw<<<NE, 128, 0, stream>>>(rbff, W_cbf3, rWb);
  k_cbf<<<nblk(NT3, 256), 256, 0, stream>>>(vecb, ddb, rWb, id_ba, id_ca, cbfb);

  // edge embedding: m = silu([h_src, h_tgt, rbf] @ W_edge + b)
  k_concat<<<nblk((long)NE * 384, 256), 256, 0, stream>>>(hbf, rbfb, e_src, e_tgt, Acat, RDIM);
  gemm_launch(stream, Acat, wtEdge, b_edge, nullptr, nullptr, 0.f, mf, mb, NE, EMDIM, 384, 1);

  // ---- interaction layers ----
  for (int l = 0; l < LNUM; ++l) {
    // rbf3e = rbf @ fused3
    gemm_launch(stream, rbfb, wtF3 + (long)l * 256 * 128, nullptr, nullptr, nullptr, 0.f,
                gate, nullptr, NE, EMDIM, RDIM, 0);
    // X1 = silu(m @ W_ba) * rbf3e
    gemm_launch(stream, mb, wtBa + (long)l * 256 * 256, nullptr, gate, nullptr, 0.f,
                nullptr, X1b, NE, EMDIM, EMDIM, 1 | 2);
    // x_trip = silu(X1 @ W_downtrip)
    gemm_launch(stream, X1b, wtDt + (long)l * 32 * 256, nullptr, nullptr, nullptr, 0.f,
                nullptr, xtb, NE, TRDIM, EMDIM, 1);
    // triplet bilinear -> x3
    k_zero<<<nblk((long)NE * BILD, 256), 256, 0, stream>>>(x3f, (long)NE * BILD);
    k_trip<<<NT3 / 128, 256, 0, stream>>>(xtb, cbfb, id_ba, id_ca,
                                          wtBil + (long)l * CDIM * BILD * TRDIM, x3f);
    // m = (m + silu(x3 @ W_up)) * isq2
    k_f2b<<<nblk((long)NE * BILD, 256), 256, 0, stream>>>(x3f, x3b, (long)NE * BILD);
    gemm_launch(stream, x3b, wtUp + (long)l * 256 * 64, nullptr, nullptr, mf, isq2,
                mf, mb, NE, EMDIM, BILD, 1 | 4);
    // residual MLP
    gemm_launch(stream, mb, wtR1 + (long)l * 256 * 256, nullptr, nullptr, nullptr, 0.f,
                nullptr, X1b, NE, EMDIM, EMDIM, 1);
    gemm_launch(stream, X1b, wtR2 + (long)l * 256 * 256, nullptr, nullptr, mf, isq2,
                mf, mb, NE, EMDIM, EMDIM, 1 | 4);
    // atom update: ha = seg(m * (rbf@fused_h), tgt) @ W_atom
    gemm_launch(stream, rbfb, wtFh + (long)l * 256 * 128, nullptr, nullptr, nullptr, 0.f,
                gate, nullptr, NE, EMDIM, RDIM, 0);
    k_zero<<<nblk((long)NA * EMDIM, 256), 256, 0, stream>>>(Hm, (long)NA * EMDIM);
    k_scatter<<<nblk((long)NE * EMDIM, 256), 256, 0, stream>>>(mf, gate, e_tgt, Hm);
    k_f2b<<<nblk((long)NA * EMDIM, 256), 256, 0, stream>>>(Hm, Hmb, (long)NA * EMDIM);
    gemm_launch(stream, Hmb, wtAt + (long)l * 128 * 256, nullptr, nullptr, nullptr, 0.f,
                haf, nullptr, NA, ADIM, EMDIM, 0);
    // Ewald block
    k_sf<<<BMOL * KPTS, 128, 0, stream>>>(cosk, sink, hf, kff, sfre, sfim);
    k_ew<<<NA, 128, 0, stream>>>(cosk, sink, sfre, sfim, ewf);
    k_f2b<<<nblk((long)NA * ADIM, 256), 256, 0, stream>>>(ewf, ewb, (long)NA * ADIM);
    gemm_launch(stream, ewb, wtEwW + (long)l * 128 * 128, nullptr, nullptr, nullptr, 0.f,
                ewWf, nullptr, NA, ADIM, ADIM, 0);
    k_hup<<<nblk((long)NA * ADIM, 256), 256, 0, stream>>>(haf, ewWf, hf, hbf);
    // edge refresh
    k_concat<<<nblk((long)NE * 512, 256), 256, 0, stream>>>(hbf, mb, e_src, e_tgt, Acat, EMDIM);
    gemm_launch(stream, Acat, wtCat + (long)l * 256 * 512, nullptr, nullptr, nullptr, 0.f,
                mf, mb, NE, EMDIM, 512, 1);
  }

  // ---- output block ----
  gemm_launch(stream, rbfb, wtFo, nullptr, nullptr, nullptr, 0.f, gate, nullptr, NE, EMDIM, RDIM, 0);
  k_zero<<<nblk((long)NA * EMDIM, 256), 256, 0, stream>>>(Hm, (long)NA * EMDIM);
  k_scatter<<<nblk((long)NE * EMDIM, 256), 256, 0, stream>>>(mf, gate, e_tgt, Hm);
  k_f2b<<<nblk((long)NA * EMDIM, 256), 256, 0, stream>>>(Hm, Hmb, (long)NA * EMDIM);
  gemm_launch(stream, Hmb, wtO1, nullptr, nullptr, nullptr, 0.f, qf, nullptr, NA, ADIM, EMDIM, 1);
  k_final<<<BMOL, 256, 0, stream>>>(qf, W_out2, out);
}
